// DoorLocalModel_57363583205572
// MI455X (gfx1250) — compile-verified, matrix operand order checked by reading, not run on hardware
//
#include <hip/hip_runtime.h>

typedef unsigned short u16;
typedef unsigned char  u8;
typedef __attribute__((ext_vector_type(16))) __bf16 v16bf;
typedef __attribute__((ext_vector_type(8)))  float  v8f;

#define NENV   1024
#define DPG    49152
#define NDOOR  (4*DPG)
#define FEAT   324
#define FEATP  352
#define ND_    600
#define SVN    728
#define SVNP   768

__device__ __forceinline__ u16 f2bf(float f) {
    unsigned u = __float_as_uint(f);
    unsigned r = u + 0x7FFFu + ((u >> 16) & 1u);   // round-to-nearest-even
    return (u16)(r >> 16);
}
__device__ __forceinline__ float selu_f(float x) {
    const float a = 1.6732632423543772f, s = 1.0507009873554805f;
    return x > 0.f ? s * x : s * a * (__expf(x) - 1.f);
}

// ISA §10.2: for addresses in the LDS aperture, the LDS address is the low 32 bits of the address, so the
// low 32 bits of a generic pointer to __shared__ are the base-relative LDS
// byte address expected by GLOBAL_LOAD_ASYNC_TO_LDS's VDST operand.
__device__ __forceinline__ uint32_t lds_addr(const u16* p) {
    return (uint32_t)(uintptr_t)p;
}
// CDNA5 async memory->LDS (ASYNCcnt-tracked), GV addressing mode.
__device__ __forceinline__ void ald_b128(uint32_t loff, const u16* g) {
    asm volatile("global_load_async_to_lds_b128 %0, %1, off"
                 :: "v"(loff), "v"(g) : "memory");
}
__device__ __forceinline__ void wait_async0() {
    asm volatile("s_wait_asynccnt 0x0" ::: "memory");
}

// ---------------------------------------------------------------- weights ->
// src is row-major (K x Nsrc) fp32; dst is (Npad x Kpad) bf16, zero padded.
__global__ void convert_wt(const float* __restrict__ src, u16* __restrict__ dst,
                           int K, int Kpad, int Nsrc, int Nreal, int Npad) {
    int id = blockIdx.x * blockDim.x + threadIdx.x;
    if (id >= Npad * Kpad) return;
    int n = id / Kpad, k = id - n * Kpad;
    float v = (k < K && n < Nreal) ? src[(size_t)k * Nsrc + n] : 0.f;
    dst[(size_t)n * Kpad + k] = f2bf(v);
}

// ------------------------------------------------------------- patch gather
__global__ void gather_patches(const float* __restrict__ map_grid,
                               const int* __restrict__ door_env,
                               const int* __restrict__ door_x,
                               const int* __restrict__ door_y,
                               u16* __restrict__ Apatch) {
    int gid = blockIdx.x * blockDim.x + threadIdx.x;
    if (gid >= NDOOR * 88) return;
    int d = gid / 88;
    int s = gid - d * 88;            // slot: 81 (kx,ky) pairs + 7 zero-pad slots
    union { u16 s4[4]; uint2 v; } o;
    o.v = make_uint2(0u, 0u);
    if (s < 81) {
        int kx = s / 9, ky = s - kx * 9;
        int env = door_env[d];
        int x = door_x[d] - 4 + kx; x = min(max(x, 0), 71);
        int y = door_y[d] - 4 + ky; y = min(max(y, 0), 71);
        const float* base = map_grid + (size_t)env * (4 * 5184) + x * 72 + y;
#pragma unroll
        for (int c = 0; c < 4; c++) o.s4[c] = f2bf(base[c * 5184]);
    }
    *(uint2*)&Apatch[(size_t)d * FEATP + s * 4] = o.v;
}

// ----------------------------------------------------------------- gin pack
__global__ void pack_gin(const u8* __restrict__ rm, const float* __restrict__ steps,
                         u16* __restrict__ gin) {
    int id = blockIdx.x * blockDim.x + threadIdx.x;
    if (id >= NENV * 288) return;
    int n = id / 288, c = id - n * 288;
    float v = (c < 256) ? (float)rm[n * 256 + c] : (c == 256 ? steps[n] : 0.f);
    gin[id] = f2bf(v);
}

// ---------------------------------------------------------------- comb pack
__global__ void pack_comb(const u16* __restrict__ localBuf, const u16* __restrict__ gbf,
                          const int* __restrict__ env, u16* __restrict__ comb) {
    int id = blockIdx.x * blockDim.x + threadIdx.x;   // one 8-bf16 chunk each
    if (id >= NDOOR * 64) return;
    int m = id >> 6, c = id & 63;
    uint4 v;
    if (c < 32) v = *(const uint4*)&localBuf[(size_t)m * 256 + c * 8];
    else        v = *(const uint4*)&gbf[(size_t)env[m] * 256 + (c - 32) * 8];
    *(uint4*)&comb[(size_t)m * 512 + c * 8] = v;
}

__global__ void f32_to_bf16_k(const float* __restrict__ src, u16* __restrict__ dst, int n) {
    int id = blockIdx.x * blockDim.x + threadIdx.x;
    if (id < n) dst[id] = f2bf(src[id]);
}

// ------------------------------------------------------------ WMMA GEMM core
// C[M,N] = selu?(A[M,K] * Bt[N,K]^T + bias), A/Bt bf16. K mult 32, N mult 128,
// M mult 128. Block tile 128x128, 8 waves (4M x 2N), 32x64 per wave = 8 WMMA
// per 32-K step. Double-buffered LDS staged with global_load_async_to_lds.
// MODE 0: bf16 out (+SELU). MODE 1: comb round -> cols<256 to localOut bf16,
// cols>=256 atomic segment-sum into gsum[env[row]]. MODE 2: fp32 out, no SELU.
template <int MODE>
__global__ void __launch_bounds__(256)
gemm_wmma(const u16* __restrict__ A, int lda,
          const u16* __restrict__ Bt,
          const float* __restrict__ bias, int Nreal, int K,
          u16* __restrict__ outb, int ldo,
          float* __restrict__ outf,
          u16* __restrict__ localOut,
          float* __restrict__ gsum,
          const int* __restrict__ env) {
    __shared__ u16 lA[2][128 * 32];
    __shared__ u16 lB[2][128 * 32];
    const int tid = threadIdx.x;
    const int lane = tid & 31, wid = tid >> 5;
    const int waveM = wid & 3, waveN = wid >> 2;
    const size_t rowBase = (size_t)blockIdx.x * 128;
    const int    colBase = blockIdx.y * 128;

    // this thread's four 16B staging chunks per k-step (rows r and r+64)
    const int r = tid >> 2, seg = tid & 3;
    const u16* gA0 = A + (rowBase + r) * (size_t)lda + seg * 8;
    const u16* gA1 = gA0 + (size_t)64 * lda;
    const u16* gB0 = Bt + ((size_t)colBase + r) * lda + seg * 8;
    const u16* gB1 = gB0 + (size_t)64 * lda;
    uint32_t loA0[2], loA1[2], loB0[2], loB1[2];
#pragma unroll
    for (int b = 0; b < 2; b++) {
        loA0[b] = lds_addr(&lA[b][r * 32 + seg * 8]);
        loA1[b] = lds_addr(&lA[b][(r + 64) * 32 + seg * 8]);
        loB0[b] = lds_addr(&lB[b][r * 32 + seg * 8]);
        loB1[b] = lds_addr(&lB[b][(r + 64) * 32 + seg * 8]);
    }

    v8f acc[2][4] = {};
    union Frag { uint4 q[2]; v16bf v; };

    const int nsteps = K >> 5;
    // prologue: stage k-step 0 into buffer 0
    ald_b128(loA0[0], gA0); ald_b128(loA1[0], gA1);
    ald_b128(loB0[0], gB0); ald_b128(loB1[0], gB1);
    gA0 += 32; gA1 += 32; gB0 += 32; gB1 += 32;

    for (int s = 0; s < nsteps; ++s) {
        const int buf = s & 1;
        wait_async0();        // my writes into `buf` have landed in LDS
        __syncthreads();      // everyone's writes landed; prior reads of buf^1 done
        if (s + 1 < nsteps) { // stage next step behind this step's WMMAs
            const int nb = buf ^ 1;
            ald_b128(loA0[nb], gA0); ald_b128(loA1[nb], gA1);
            ald_b128(loB0[nb], gB0); ald_b128(loB1[nb], gB1);
            gA0 += 32; gA1 += 32; gB0 += 32; gB1 += 32;
        }
        const u16* cA = lA[buf];
        const u16* cB = lB[buf];
        // per-lane fragments per ISA 7.12.2 16-bit layout:
        // lane(hi=lane>>4) holds k in [hi*8, hi*8+8) and [16+hi*8, 16+hi*8+8)
        Frag fa[2], fb[4];
        const int hi = (lane >> 4) * 8;
#pragma unroll
        for (int i = 0; i < 2; i++) {
            int m = waveM * 32 + i * 16 + (lane & 15);
            fa[i].q[0] = *(const uint4*)&cA[m * 32 + hi];
            fa[i].q[1] = *(const uint4*)&cA[m * 32 + 16 + hi];
        }
#pragma unroll
        for (int i = 0; i < 4; i++) {
            int n = waveN * 64 + i * 16 + (lane & 15);
            fb[i].q[0] = *(const uint4*)&cB[n * 32 + hi];
            fb[i].q[1] = *(const uint4*)&cB[n * 32 + 16 + hi];
        }
#pragma unroll
        for (int mi = 0; mi < 2; mi++)
#pragma unroll
            for (int ni = 0; ni < 4; ni++)
                acc[mi][ni] = __builtin_amdgcn_wmma_f32_16x16x32_bf16(
                    false, fa[mi].v, false, fb[ni].v, (short)0, acc[mi][ni],
                    false, false);
    }

    // epilogue: D layout — lane&15 = col, (lane>>4)*8 + vgpr = row-in-tile
#pragma unroll
    for (int mi = 0; mi < 2; mi++) {
#pragma unroll
        for (int v = 0; v < 8; v++) {
            size_t row = rowBase + waveM * 32 + mi * 16 + ((lane >> 4) << 3) + v;
            int e = (MODE == 1) ? env[row] : 0;
#pragma unroll
            for (int ni = 0; ni < 4; ni++) {
                int col = colBase + waveN * 64 + ni * 16 + (lane & 15);
                float b = (MODE == 2) ? (col < Nreal ? bias[col] : 0.f) : bias[col];
                float val = acc[mi][ni][v] + b;
                if (MODE == 0) {
                    outb[row * (size_t)ldo + col] = f2bf(selu_f(val));
                } else if (MODE == 1) {
                    float sv = selu_f(val);
                    if (col < 256)
                        localOut[row * 256 + col] = f2bf(sv);
                    else
                        atomicAdd(&gsum[(size_t)e * 256 + (col - 256)], sv);
                } else {
                    if (col < Nreal) outf[row * (size_t)ldo + col] = val;
                }
            }
        }
    }
}

// ------------------------------------------------------------- final reduce
__global__ void finalize(const float* __restrict__ logodds, const u8* __restrict__ dc,
                         float* __restrict__ out) {
    int gw = (blockIdx.x * blockDim.x + threadIdx.x) >> 5;   // one wave32 per env
    int lane = threadIdx.x & 31;
    if (gw >= NENV) return;
    const float* lo = logodds + (size_t)gw * SVNP;
    const u8* dcr = dc + (size_t)gw * ND_;
    float sum = 0.f;
    for (int j = lane; j < SVN; j += 32) {
        float p = (j < ND_ && dcr[j]) ? 1.f : 1.f / (1.f + __expf(-lo[j]));
        sum += p;
    }
#pragma unroll
    for (int off = 16; off; off >>= 1) sum += __shfl_xor(sum, off, 32);
    if (lane == 0) out[gw] = sum * 0.5f;
}

// ================================================================== launch
extern "C" void kernel_launch(void* const* d_in, const int* in_sizes, int n_in,
                              void* d_out, int out_size, void* d_ws, size_t ws_size,
                              hipStream_t stream) {
    const float* map_grid = (const float*)d_in[0];
    const u8*    room_mask = (const u8*)d_in[1];
    const float* steps = (const float*)d_in[2];
    const int*   door_env = (const int*)d_in[3];
    const int*   door_x = (const int*)d_in[4];
    const int*   door_y = (const int*)d_in[5];
    const u8*    door_connects = (const u8*)d_in[6];

    // params: insertion order vs jax sorted-key order, detect by size signature
    const float *doorW, *doorB, *globalW, *globalB, *locW0, *locW1, *locB0, *locB1,
                *fcW0, *fcW1, *fcB0, *fcB1, *svW, *svB;
    bool sorted = (n_in > 9) && (in_sizes[9] == 131072);
    if (!sorted) {
        doorW = (const float*)d_in[7];  doorB = (const float*)d_in[8];
        globalW = (const float*)d_in[9]; globalB = (const float*)d_in[10];
        locW0 = (const float*)d_in[11]; locW1 = (const float*)d_in[12];
        locB0 = (const float*)d_in[13]; locB1 = (const float*)d_in[14];
        fcW0 = (const float*)d_in[15];  fcW1 = (const float*)d_in[16];
        fcB0 = (const float*)d_in[17];  fcB1 = (const float*)d_in[18];
        svW = (const float*)d_in[19];   svB = (const float*)d_in[20];
    } else {
        doorW = (const float*)d_in[7];  doorB = (const float*)d_in[8];
        fcW0 = (const float*)d_in[9];   fcW1 = (const float*)d_in[10];
        fcB0 = (const float*)d_in[11];  fcB1 = (const float*)d_in[12];
        globalW = (const float*)d_in[13]; globalB = (const float*)d_in[14];
        locW0 = (const float*)d_in[15]; locW1 = (const float*)d_in[16];
        locB0 = (const float*)d_in[17]; locB1 = (const float*)d_in[18];
        svW = (const float*)d_in[19];   svB = (const float*)d_in[20];
    }

    // ---- workspace carve (≈310 MB; patch buffer aliased with comb buffer)
    char* p = (char*)d_ws;
    auto carve = [&](size_t bytes) -> char* {
        char* r = p; p += (bytes + 255) & ~(size_t)255; return r;
    };
    u16*   region0 = (u16*)carve((size_t)NDOOR * 512 * 2);  // Apatch(352) then comb(512)
    u16*   localB  = (u16*)carve((size_t)NDOOR * 256 * 2);
    u16*   gbf     = (u16*)carve((size_t)NENV * 256 * 2);
    float* gsum    = (float*)carve((size_t)NENV * 256 * 4);
    u16*   gin     = (u16*)carve((size_t)NENV * 288 * 2);
    u16*   fcb0    = (u16*)carve((size_t)NENV * 512 * 2);
    u16*   fcb1    = (u16*)carve((size_t)NENV * 512 * 2);
    float* logodds = (float*)carve((size_t)NENV * SVNP * 4);
    u16*   doorWt  = (u16*)carve((size_t)4 * 256 * FEATP * 2);
    u16*   glbWt   = (u16*)carve((size_t)256 * 288 * 2);
    u16*   locWt0  = (u16*)carve((size_t)512 * 512 * 2);
    u16*   locWt1  = (u16*)carve((size_t)512 * 512 * 2);
    u16*   fcWt0   = (u16*)carve((size_t)512 * 256 * 2);
    u16*   fcWt1   = (u16*)carve((size_t)512 * 512 * 2);
    u16*   svWt    = (u16*)carve((size_t)SVNP * 512 * 2);
    (void)ws_size; (void)out_size;

    auto cvt = [&](const float* src, u16* dst, int K, int Kpad, int Nsrc, int Nreal, int Npad) {
        int total = Npad * Kpad;
        convert_wt<<<(total + 255) / 256, 256, 0, stream>>>(src, dst, K, Kpad, Nsrc, Nreal, Npad);
    };
    for (int g = 0; g < 4; g++)
        cvt(doorW + (size_t)g * FEAT * 256, doorWt + (size_t)g * 256 * FEATP,
            FEAT, FEATP, 256, 256, 256);
    cvt(globalW, glbWt, 257, 288, 256, 256, 256);
    cvt(locW0, locWt0, 512, 512, 512, 512, 512);
    cvt(locW1, locWt1, 512, 512, 512, 512, 512);
    cvt(fcW0, fcWt0, 256, 256, 512, 512, 512);
    cvt(fcW1, fcWt1, 512, 512, 512, 512, 512);
    cvt(svW, svWt, 512, 512, SVN, SVN, SVNP);

    auto gemm0 = [&](const u16* A, int lda, const u16* Bt, const float* bias,
                     int M, int N, u16* outb, int ldo) {
        dim3 grid(M / 128, N / 128);
        gemm_wmma<0><<<grid, 256, 0, stream>>>(A, lda, Bt, bias, N, lda, outb, ldo,
                                               nullptr, nullptr, nullptr, nullptr);
    };

    // global MLP: g0 = selu(gin @ global_W + b) -> bf16
    pack_gin<<<(NENV * 288 + 255) / 256, 256, 0, stream>>>(room_mask, steps, gin);
    gemm0(gin, 288, glbWt, globalB, NENV, 256, gbf, 256);

    // patch gather + per-group door GEMMs -> local bf16
    gather_patches<<<((size_t)NDOOR * 88 + 255) / 256, 256, 0, stream>>>(
        map_grid, door_env, door_x, door_y, region0);
    for (int g = 0; g < 4; g++)
        gemm0(region0 + (size_t)g * DPG * FEATP, FEATP,
              doorWt + (size_t)g * 256 * FEATP, doorB + (size_t)g * 256,
              DPG, 256, localB + (size_t)g * DPG * 256, 256);

    // two message-passing rounds: comb = [local | g[env]]; GEMM 512->512;
    // cols<256 -> local, cols>=256 -> L2-resident atomic segment-sum
    for (int r = 0; r < 2; r++) {
        hipMemsetAsync(gsum, 0, (size_t)NENV * 256 * 4, stream);
        pack_comb<<<((size_t)NDOOR * 64 + 255) / 256, 256, 0, stream>>>(
            localB, gbf, door_env, region0);
        dim3 grid(NDOOR / 128, 512 / 128);
        gemm_wmma<1><<<grid, 256, 0, stream>>>(
            region0, 512, r ? locWt1 : locWt0, r ? locB1 : locB0, 512, 512,
            nullptr, 0, nullptr, localB, gsum, door_env);
        f32_to_bf16_k<<<(NENV * 256 + 255) / 256, 256, 0, stream>>>(gsum, gbf, NENV * 256);
    }

    // FC stack + state-value head
    gemm0(gbf, 256, fcWt0, fcB0, NENV, 512, fcb0, 512);
    gemm0(fcb0, 512, fcWt1, fcB1, NENV, 512, fcb1, 512);
    {
        dim3 grid(NENV / 128, SVNP / 128);
        gemm_wmma<2><<<grid, 256, 0, stream>>>(fcb1, 512, svWt, svB, SVN, 512,
                                               nullptr, SVNP, logodds,
                                               nullptr, nullptr, nullptr);
    }
    finalize<<<(NENV * 32 + 255) / 256, 256, 0, stream>>>(logodds, door_connects,
                                                          (float*)d_out);
}
